// Het_GNN_86706799771648
// MI455X (gfx1250) — matrix operand order repaired and checked by reading, not compiled, not run on hardware
//
#include <hip/hip_runtime.h>
#include <hip/hip_bf16.h>

#define DEV __device__ __forceinline__

typedef __attribute__((ext_vector_type(16))) __bf16 v16bf;
typedef __attribute__((ext_vector_type(8)))  float  v8f;

union FragU { uint4 q[2]; v16bf v; };

DEV float sigmoid_f(float x) { return 1.0f / (1.0f + __expf(-x)); }

DEV unsigned short f2bf(float f) {  // round-half-up (epilogue use only)
  unsigned int u = __float_as_uint(f) + 0x8000u;
  return (unsigned short)(u >> 16);
}

DEV uint4 zero4() { uint4 z; z.x = 0u; z.y = 0u; z.z = 0u; z.w = 0u; return z; }

// ---------------------------------------------------------------------------
// Packed fp32 -> bf16 conversion (2 elements/dword out): 3 VALU per 2 elements.
// ---------------------------------------------------------------------------
__global__ void cvt_f32_bf16_kernel(const float* __restrict__ src,
                                    unsigned int* __restrict__ dst, long long n_pairs) {
  long long i = (long long)blockIdx.x * blockDim.x + threadIdx.x;
  long long stride = (long long)gridDim.x * blockDim.x;
  for (; i < n_pairs; i += stride) {
    float2 f = reinterpret_cast<const float2*>(src)[i];
    unsigned int a = __float_as_uint(f.x) + 0x8000u;
    unsigned int b = __float_as_uint(f.y) + 0x8000u;
    // result bytes [1:0] = a[3:2], [3:2] = b[3:2]  -> two packed bf16
    dst[i] = __builtin_amdgcn_perm(b, a, 0x07060302u);
  }
}

// ---------------------------------------------------------------------------
// Fragment loaders: pure 128-bit loads bit-cast into the WMMA operand (no VALU).
// A (16x32 MxK): lane<16 -> row M=lane, K {0..7,16..23}; lane>=16 -> {8..15,24..31}.
// B (32x16 KxN) from weights stored [Nrows, stride] row-major (B[k,n]=W[n,k]):
//   col n = lane&15; lane<16 -> K {0..7,16..23}, lane>=16 -> K {8..15,24..31}.
// ---------------------------------------------------------------------------
template<bool CHK>
DEV v16bf load_a_bf(const unsigned short* __restrict__ A, int r0, int stride, int k0, int K) {
  int lane = threadIdx.x & 31;
  long long ro = (long long)(r0 + (lane & 15)) * stride;
  int kb = k0 + ((lane & 16) ? 8 : 0);
  FragU f;
  if (CHK) {
    f.q[0] = (kb + 8  <= K) ? *reinterpret_cast<const uint4*>(A + ro + kb)      : zero4();
    f.q[1] = (kb + 24 <= K) ? *reinterpret_cast<const uint4*>(A + ro + kb + 16) : zero4();
  } else {
    f.q[0] = *reinterpret_cast<const uint4*>(A + ro + kb);
    f.q[1] = *reinterpret_cast<const uint4*>(A + ro + kb + 16);
  }
  return f.v;
}

template<bool CHK>
DEV v16bf load_b_bf(const unsigned short* __restrict__ W, int rowbase, int n0, int nmax,
                    int stride, int k0, int K) {
  int lane = threadIdx.x & 31;
  int n = n0 + (lane & 15);
  if (n > nmax - 1) n = nmax - 1;
  long long ro = (long long)(rowbase + n) * stride;
  int ks = k0 + ((lane & 16) ? 8 : 0);
  FragU f;
  if (CHK) {
    f.q[0] = (ks + 8  <= K) ? *reinterpret_cast<const uint4*>(W + ro + ks)     : zero4();
    f.q[1] = (ks + 24 <= K) ? *reinterpret_cast<const uint4*>(W + ro + ks + 16) : zero4();
  } else {
    f.q[0] = *reinterpret_cast<const uint4*>(W + ro + ks);
    f.q[1] = *reinterpret_cast<const uint4*>(W + ro + ks + 16);
  }
  return f.v;
}

DEV v8f wmma_bf16(v16bf a, v16bf b, v8f c) {
  return __builtin_amdgcn_wmma_f32_16x16x32_bf16(false, a, false, b, (short)0, c, false, false);
}

DEV v8f vzero8() { v8f z = {0.f, 0.f, 0.f, 0.f, 0.f, 0.f, 0.f, 0.f}; return z; }

// ---------------------------------------------------------------------------
// C[R,N] = A[R,K] * W[N,K]^T + bias.  A, W bf16; f32 accumulate.
// Block = 128 threads = 4 waves. Wave computes 32x64 (8 wmma accumulators).
// ---------------------------------------------------------------------------
template<int C_BF, bool KCHK>
__global__ void gemm_bias_kernel(const unsigned short* __restrict__ A,
                                 const unsigned short* __restrict__ W,
                                 const float* __restrict__ bias, void* __restrict__ C,
                                 int R, int N, int K) {
  (void)R;
  int wave = threadIdx.x >> 5;
  int lane = threadIdx.x & 31;
  int r0 = blockIdx.y * 32;
  int n_wave = blockIdx.x * 256 + wave * 64;
  if (n_wave >= N) return;  // uniform per wave

  v8f acc[2][4];
#pragma unroll
  for (int h = 0; h < 2; ++h)
#pragma unroll
    for (int j = 0; j < 4; ++j) acc[h][j] = vzero8();

  int Kp = (K + 31) & ~31;
  for (int k0 = 0; k0 < Kp; k0 += 32) {
    v16bf a0 = load_a_bf<KCHK>(A, r0,      K, k0, K);
    v16bf a1 = load_a_bf<KCHK>(A, r0 + 16, K, k0, K);
#pragma unroll
    for (int j = 0; j < 4; ++j) {
      v16bf bf = load_b_bf<KCHK>(W, 0, n_wave + j * 16, N, K, k0, K);
      acc[0][j] = wmma_bf16(a0, bf, acc[0][j]);
      acc[1][j] = wmma_bf16(a1, bf, acc[1][j]);
    }
  }

  // D layout: lane<16 -> N=lane, M=v; lane>=16 -> N=lane-16, M=8+v
  int m_base = (lane & 16) ? 8 : 0;
  int nl = lane & 15;
#pragma unroll
  for (int h = 0; h < 2; ++h) {
#pragma unroll
    for (int j = 0; j < 4; ++j) {
      int n = n_wave + j * 16 + nl;
      if (n < N) {
        float bv = bias[n];
#pragma unroll
        for (int v = 0; v < 8; ++v) {
          long long row = r0 + h * 16 + m_base + v;
          float val = acc[h][j][v] + bv;
          if (C_BF) ((unsigned short*)C)[row * N + n] = f2bf(val);
          else      ((float*)C)[row * N + n] = val;
        }
      }
    }
  }
}

// ---------------------------------------------------------------------------
// Fused single-step BiLSTM (h0=c0=0 => forget gate dead):
//   h = sig(o)*tanh( sig(i)*tanh(g) ),  gates = Y @ Wl[dir]^T + bl[dir]
// Y [R,512] bf16; Wl bf16 flat [8000,512] (2 dirs x [i f g o] x 1000); bl f32 [8000].
// Optional pair-average with tmp_in; writes f32 and/or bf16 [R,2000].
// Block = 128 = 4 waves; wave: one (dir, 16-col) tile x 32 rows, 6 accumulators.
// ---------------------------------------------------------------------------
__global__ void lstm_fused_kernel(const unsigned short* __restrict__ Y,
                                  const unsigned short* __restrict__ Wl,
                                  const float* __restrict__ bl,
                                  const float* __restrict__ tmp_in,
                                  float* __restrict__ out_f32,
                                  unsigned short* __restrict__ out_bf16,
                                  int R) {
  (void)R;
  int wave = threadIdx.x >> 5;
  int lane = threadIdx.x & 31;
  int kt = blockIdx.x * 4 + wave;       // 0..127; 63 col-tiles per direction
  if (kt >= 126) return;                // uniform per wave
  int dir = kt / 63;
  int n0 = (kt % 63) * 16;
  int r0 = blockIdx.y * 32;

  v8f acc[2][3];
#pragma unroll
  for (int h = 0; h < 2; ++h)
#pragma unroll
    for (int g = 0; g < 3; ++g) acc[h][g] = vzero8();

  int gbase[3];
  gbase[0] = dir * 4000;          // i
  gbase[1] = dir * 4000 + 2000;   // g
  gbase[2] = dir * 4000 + 3000;   // o

  for (int k0 = 0; k0 < 512; k0 += 32) {
    v16bf a0 = load_a_bf<false>(Y, r0,      512, k0, 512);
    v16bf a1 = load_a_bf<false>(Y, r0 + 16, 512, k0, 512);
#pragma unroll
    for (int g = 0; g < 3; ++g) {
      v16bf bf = load_b_bf<false>(Wl, gbase[g], n0, 1000, 512, k0, 512);
      acc[0][g] = wmma_bf16(a0, bf, acc[0][g]);
      acc[1][g] = wmma_bf16(a1, bf, acc[1][g]);
    }
  }

  int m_base = (lane & 16) ? 8 : 0;
  int n = n0 + (lane & 15);
  if (n < 1000) {
    float b_i = bl[gbase[0] + n];
    float b_g = bl[gbase[1] + n];
    float b_o = bl[gbase[2] + n];
    int col = dir * 1000 + n;
#pragma unroll
    for (int h = 0; h < 2; ++h) {
#pragma unroll
      for (int v = 0; v < 8; ++v) {
        long long row = r0 + h * 16 + m_base + v;
        float gi = acc[h][0][v] + b_i;
        float gg = acc[h][1][v] + b_g;
        float go = acc[h][2][v] + b_o;
        float cc = sigmoid_f(gi) * tanhf(gg);
        float hh = sigmoid_f(go) * tanhf(cc);
        long long off = row * 2000 + col;
        if (tmp_in)   hh = 0.5f * (hh + tmp_in[off]);
        if (out_f32)  out_f32[off] = hh;
        if (out_bf16) out_bf16[off] = f2bf(hh);
      }
    }
  }
}

// mean over 24 neighbors: O [512*24, 2000] -> out [512, 2000]
__global__ void mean24_kernel(const float* __restrict__ O, float* __restrict__ out) {
  long long idx = (long long)blockIdx.x * blockDim.x + threadIdx.x;
  if (idx >= 512LL * 2000LL) return;
  long long b = idx / 2000;
  long long e = idx % 2000;
  const float* base = O + b * 24 * 2000 + e;
  float s = 0.f;
#pragma unroll
  for (int n = 0; n < 24; ++n) s += base[n * 2000];
  out[idx] = s * (1.0f / 24.0f);
}

// attention over {c, u_agg, p_agg} with p_att [4000], leaky_relu(0.01), softmax, combine
__global__ void attn_combine_kernel(const float* __restrict__ c, const float* __restrict__ u,
                                    const float* __restrict__ p, const float* __restrict__ att,
                                    float* __restrict__ out) {
  int b = blockIdx.x;
  int t = threadIdx.x;
  const float* cb = c + (long long)b * 2000;
  const float* ub = u + (long long)b * 2000;
  const float* pb = p + (long long)b * 2000;
  float s0 = 0.f, s1 = 0.f, s2 = 0.f, s3 = 0.f;
  for (int e = t; e < 2000; e += 256) {
    float al = att[e], ah = att[2000 + e];
    float cv = cb[e];
    s0 += cv * al;
    s1 += cv * ah;
    s2 += ub[e] * ah;
    s3 += pb[e] * ah;
  }
  __shared__ float red[4][256];
  red[0][t] = s0; red[1][t] = s1; red[2][t] = s2; red[3][t] = s3;
  __syncthreads();
  for (int off = 128; off > 0; off >>= 1) {
    if (t < off) {
#pragma unroll
      for (int q = 0; q < 4; ++q) red[q][t] += red[q][t + off];
    }
    __syncthreads();
  }
  __shared__ float w[3];
  if (t == 0) {
    float a0 = red[0][0] + red[1][0];
    float a1 = red[0][0] + red[2][0];
    float a2 = red[0][0] + red[3][0];
    a0 = (a0 > 0.f) ? a0 : 0.01f * a0;
    a1 = (a1 > 0.f) ? a1 : 0.01f * a1;
    a2 = (a2 > 0.f) ? a2 : 0.01f * a2;
    float mx = fmaxf(a0, fmaxf(a1, a2));
    float e0 = __expf(a0 - mx), e1 = __expf(a1 - mx), e2 = __expf(a2 - mx);
    float inv = 1.0f / (e0 + e1 + e2);
    w[0] = e0 * inv; w[1] = e1 * inv; w[2] = e2 * inv;
  }
  __syncthreads();
  float w0 = w[0], w1 = w[1], w2 = w[2];
  for (int e = t; e < 2000; e += 256)
    out[(long long)b * 2000 + e] = w0 * cb[e] + w1 * ub[e] + w2 * pb[e];
}

// ---------------------------------------------------------------------------
extern "C" void kernel_launch(void* const* d_in, const int* in_sizes, int n_in,
                              void* d_out, int out_size, void* d_ws, size_t ws_size,
                              hipStream_t stream) {
  (void)in_sizes; (void)n_in; (void)out_size; (void)ws_size;

  const float* c_text  = (const float*)d_in[0];
  const float* c_image = (const float*)d_in[1];
  const float* p_text  = (const float*)d_in[2];
  const float* p_image = (const float*)d_in[3];
  const float* u_text  = (const float*)d_in[4];
  const float* u_other = (const float*)d_in[5];
  const float* W_lt = (const float*)d_in[6];  const float* b_lt = (const float*)d_in[7];
  const float* W_li = (const float*)d_in[8];  const float* b_li = (const float*)d_in[9];
  const float* W_lo = (const float*)d_in[10]; const float* b_lo = (const float*)d_in[11];
  const float* Lt_W = (const float*)d_in[12]; const float* Lt_b = (const float*)d_in[13];
  const float* Li_W = (const float*)d_in[14]; const float* Li_b = (const float*)d_in[15];
  const float* Lo_W = (const float*)d_in[16]; const float* Lo_b = (const float*)d_in[17];
  const float* ua_lin_W  = (const float*)d_in[18]; const float* ua_lin_b  = (const float*)d_in[19];
  const float* ua_lstm_W = (const float*)d_in[20]; const float* ua_lstm_b = (const float*)d_in[21];
  const float* ua_out_W  = (const float*)d_in[22]; const float* ua_out_b  = (const float*)d_in[23];
  const float* pa_lin_W  = (const float*)d_in[24]; const float* pa_lin_b  = (const float*)d_in[25];
  const float* pa_lstm_W = (const float*)d_in[26]; const float* pa_lstm_b = (const float*)d_in[27];
  const float* pa_out_W  = (const float*)d_in[28]; const float* pa_out_b  = (const float*)d_in[29];
  const float* p_att     = (const float*)d_in[30];

  // ---- workspace carve-out (all 256B aligned)
  char* ws = (char*)d_ws;
  size_t off = 0;
  auto carve = [&](size_t bytes) -> void* {
    void* p = ws + off;
    off += (bytes + 255) & ~(size_t)255;
    return p;
  };
  float*          c_f32 = (float*)carve(512LL * 2000 * 4);
  float*          u_agg = (float*)carve(512LL * 2000 * 4);
  float*          p_agg = (float*)carve(512LL * 2000 * 4);
  unsigned short* Ya    = (unsigned short*)carve(12288LL * 512 * 2);
  unsigned short* Yb    = (unsigned short*)carve(12288LL * 512 * 2);
  unsigned short* Hbf   = (unsigned short*)carve(12288LL * 2000 * 2);
  unsigned short* H2    = (unsigned short*)carve(12288LL * 2000 * 2);
  unsigned short* Zbf   = (unsigned short*)carve(12288LL * 512 * 2);
  float*          Tf    = (float*)carve(12288LL * 2000 * 4);
  unsigned short* Xa    = (unsigned short*)carve(12288LL * 768 * 2);   // text inputs
  unsigned short* Xb    = (unsigned short*)carve(12288LL * 2048 * 2);  // image/other inputs
  // bf16 weights (converted once per launch; L2-resident afterwards)
  unsigned short* bWlt    = (unsigned short*)carve(512LL * 768 * 2);
  unsigned short* bWli    = (unsigned short*)carve(512LL * 2048 * 2);
  unsigned short* bWlo    = (unsigned short*)carve(512LL * 512 * 2);
  unsigned short* bLt     = (unsigned short*)carve(8000LL * 512 * 2);
  unsigned short* bLi     = (unsigned short*)carve(8000LL * 512 * 2);
  unsigned short* bLo     = (unsigned short*)carve(8000LL * 512 * 2);
  unsigned short* bUaLin  = (unsigned short*)carve(512LL * 2000 * 2);
  unsigned short* bUaLstm = (unsigned short*)carve(8000LL * 512 * 2);
  unsigned short* bUaOut  = (unsigned short*)carve(2000LL * 2000 * 2);
  unsigned short* bPaLin  = (unsigned short*)carve(512LL * 2000 * 2);
  unsigned short* bPaLstm = (unsigned short*)carve(8000LL * 512 * 2);
  unsigned short* bPaOut  = (unsigned short*)carve(2000LL * 2000 * 2);

  auto cvt = [&](const float* src, unsigned short* dst, long long n) {
    long long pairs = n / 2;
    int blocks = (int)((pairs + 255) / 256);
    if (blocks > 4096) blocks = 4096;
    cvt_f32_bf16_kernel<<<blocks, 256, 0, stream>>>(src, (unsigned int*)dst, pairs);
  };

  // ---- one-time weight conversion
  cvt(W_lt, bWlt, 512LL * 768);
  cvt(W_li, bWli, 512LL * 2048);
  cvt(W_lo, bWlo, 512LL * 512);
  cvt(Lt_W, bLt, 8000LL * 512);
  cvt(Li_W, bLi, 8000LL * 512);
  cvt(Lo_W, bLo, 8000LL * 512);
  cvt(ua_lin_W, bUaLin, 512LL * 2000);
  cvt(ua_lstm_W, bUaLstm, 8000LL * 512);
  cvt(ua_out_W, bUaOut, 2000LL * 2000);
  cvt(pa_lin_W, bPaLin, 512LL * 2000);
  cvt(pa_lstm_W, bPaLstm, 8000LL * 512);
  cvt(pa_out_W, bPaOut, 2000LL * 2000);

  const int Rc = 512, Rn = 12288;
  dim3 blk(128);

  // ---- Center (post_birnn on c_text/c_image) -> c_f32 [512,2000]
  cvt(c_text, Xa, 512LL * 768);
  cvt(c_image, Xb, 512LL * 2048);
  gemm_bias_kernel<1, false><<<dim3(2, Rc / 32), blk, 0, stream>>>(Xa, bWlt, b_lt, Ya, Rc, 512, 768);
  gemm_bias_kernel<1, false><<<dim3(2, Rc / 32), blk, 0, stream>>>(Xb, bWli, b_li, Yb, Rc, 512, 2048);
  lstm_fused_kernel<<<dim3(32, Rc / 32), blk, 0, stream>>>(Ya, bLt, Lt_b, nullptr, Tf, nullptr, Rc);
  lstm_fused_kernel<<<dim3(32, Rc / 32), blk, 0, stream>>>(Yb, bLi, Li_b, Tf, c_f32, nullptr, Rc);

  // ---- User path: user_birnn -> same_type_agg -> u_agg
  cvt(u_text, Xa, 12288LL * 768);
  cvt(u_other, Xb, 12288LL * 512);
  gemm_bias_kernel<1, false><<<dim3(2, Rn / 32), blk, 0, stream>>>(Xa, bWlt, b_lt, Ya, Rn, 512, 768);
  gemm_bias_kernel<1, false><<<dim3(2, Rn / 32), blk, 0, stream>>>(Xb, bWlo, b_lo, Yb, Rn, 512, 512);
  lstm_fused_kernel<<<dim3(32, Rn / 32), blk, 0, stream>>>(Ya, bLt, Lt_b, nullptr, Tf, nullptr, Rn);
  lstm_fused_kernel<<<dim3(32, Rn / 32), blk, 0, stream>>>(Yb, bLo, Lo_b, Tf, nullptr, Hbf, Rn);
  gemm_bias_kernel<1, true><<<dim3(2, Rn / 32), blk, 0, stream>>>(Hbf, bUaLin, ua_lin_b, Zbf, Rn, 512, 2000);
  lstm_fused_kernel<<<dim3(32, Rn / 32), blk, 0, stream>>>(Zbf, bUaLstm, ua_lstm_b, nullptr, nullptr, H2, Rn);
  gemm_bias_kernel<0, true><<<dim3(8, Rn / 32), blk, 0, stream>>>(H2, bUaOut, ua_out_b, Tf, Rn, 2000, 2000);
  mean24_kernel<<<4000, 256, 0, stream>>>(Tf, u_agg);

  // ---- Post path (reuses all large buffers) -> p_agg
  cvt(p_text, Xa, 12288LL * 768);
  cvt(p_image, Xb, 12288LL * 2048);
  gemm_bias_kernel<1, false><<<dim3(2, Rn / 32), blk, 0, stream>>>(Xa, bWlt, b_lt, Ya, Rn, 512, 768);
  gemm_bias_kernel<1, false><<<dim3(2, Rn / 32), blk, 0, stream>>>(Xb, bWli, b_li, Yb, Rn, 512, 2048);
  lstm_fused_kernel<<<dim3(32, Rn / 32), blk, 0, stream>>>(Ya, bLt, Lt_b, nullptr, Tf, nullptr, Rn);
  lstm_fused_kernel<<<dim3(32, Rn / 32), blk, 0, stream>>>(Yb, bLi, Li_b, Tf, nullptr, Hbf, Rn);
  gemm_bias_kernel<1, true><<<dim3(2, Rn / 32), blk, 0, stream>>>(Hbf, bPaLin, pa_lin_b, Zbf, Rn, 512, 2000);
  lstm_fused_kernel<<<dim3(32, Rn / 32), blk, 0, stream>>>(Zbf, bPaLstm, pa_lstm_b, nullptr, nullptr, H2, Rn);
  gemm_bias_kernel<0, true><<<dim3(8, Rn / 32), blk, 0, stream>>>(H2, bPaOut, pa_out_b, Tf, Rn, 2000, 2000);
  mean24_kernel<<<4000, 256, 0, stream>>>(Tf, p_agg);

  // ---- Attention + weighted combine -> d_out [512,2000] f32
  attn_combine_kernel<<<512, 256, 0, stream>>>(c_f32, u_agg, p_agg, p_att, (float*)d_out);
}
